// MultiHeadAttention_3109556323142
// MI455X (gfx1250) — matrix operand order repaired and checked by reading, not compiled, not run on hardware
//
#include <hip/hip_runtime.h>
#include <hip/hip_bf16.h>

// ---- problem constants (match reference) ----
#define BB 4
#define SS 2048
#define DD 1024
#define HH 16
#define DKK 64
#define MM (BB*SS)           // 8192 rows

typedef __attribute__((ext_vector_type(16))) _Float16 v16h;
typedef __attribute__((ext_vector_type(8)))  _Float16 v8h;
typedef __attribute__((ext_vector_type(8)))  float    v8f;
typedef __attribute__((ext_vector_type(4)))  float    v4f;

// ---------------- WMMA wrapper ----------------
__device__ __forceinline__ v8f wmma16(v16h a, v16h b, v8f c) {
  // D = A(16x32 f16) * B(32x16 f16) + C(16x16 f32)
  return __builtin_amdgcn_wmma_f32_16x16x32_f16(
      false, a, false, b, (short)0, c, false, false);
}

// ---------------- CDNA5 async global->LDS copies (ASYNCcnt) ----------------
// ISA 10.x / 15.18.3: GLOBAL_LOAD_ASYNC_TO_LDS_B{64,128}: per-lane copy of
// 8/16 bytes from global memory to LDS; completion tracked with ASYNCcnt.
__device__ __forceinline__ void async_copy_b128(unsigned lds_off, const void* gptr) {
  asm volatile("global_load_async_to_lds_b128 %0, %1, off"
               :: "v"(lds_off), "v"(gptr) : "memory");
}
__device__ __forceinline__ void async_copy_b64(unsigned lds_off, const void* gptr) {
  asm volatile("global_load_async_to_lds_b64 %0, %1, off"
               :: "v"(lds_off), "v"(gptr) : "memory");
}
__device__ __forceinline__ void wait_async0() {
  asm volatile("s_wait_asynccnt 0" ::: "memory");
}

// ------------- operand loaders (ISA §7.12.2 layouts) -------------
// A-matrix 16x32 f16: lane L (0-15): row M=L, K chunks {0..7} and {16..23};
// lane L+16: row M=L, K chunks {8..15} and {24..31}.
__device__ __forceinline__ v16h load_a(const float* src, int ld, int mb, int kb, int lane) {
  int hf = lane >> 4, r = lane & 15;
  const float* p0 = src + (size_t)(mb + r) * ld + kb + hf * 8;
  v4f x0 = *(const v4f*)(p0);
  v4f x1 = *(const v4f*)(p0 + 4);
  v4f y0 = *(const v4f*)(p0 + 16);
  v4f y1 = *(const v4f*)(p0 + 20);
  v16h a;
#pragma unroll
  for (int i = 0; i < 4; ++i) {
    a[i]      = (_Float16)x0[i];
    a[4 + i]  = (_Float16)x1[i];
    a[8 + i]  = (_Float16)y0[i];
    a[12 + i] = (_Float16)y1[i];
  }
  return a;
}

__device__ __forceinline__ v16h load_a(const _Float16* src, int ld, int mb, int kb, int lane) {
  int hf = lane >> 4, r = lane & 15;
  const _Float16* p0 = src + (size_t)(mb + r) * ld + kb + hf * 8;
  v8h x = *(const v8h*)(p0);        // K chunk 0 (8 halves, 16B)
  v8h y = *(const v8h*)(p0 + 16);   // K chunk 1
  v16h a;
#pragma unroll
  for (int i = 0; i < 8; ++i) { a[i] = x[i]; a[8 + i] = y[i]; }
  return a;
}

// B-matrix 32x16 f16 with B(k,n) = src[n*ld + k] (src row-major [N, K]):
// lane n (0-15) holds K chunks {0..7},{16..23} of column n; lane n+16 holds {8..15},{24..31}.
__device__ __forceinline__ v16h load_b(const _Float16* src, int ld, int kb, int nb, int lane) {
  int n = nb + (lane & 15);
  int k = kb + (lane >> 4) * 8;
  const _Float16* p = src + (size_t)n * ld + k;
  v8h x = *(const v8h*)(p);
  v8h y = *(const v8h*)(p + 16);
  v16h b;
#pragma unroll
  for (int i = 0; i < 8; ++i) { b[i] = x[i]; b[8 + i] = y[i]; }
  return b;
}

// ---------------- f32 -> f16 conversion prepass (weights) ----------------
__global__ void __launch_bounds__(256)
cvt_f32_f16(const float* __restrict__ src, _Float16* __restrict__ dst, int n8) {
  int i = blockIdx.x * blockDim.x + threadIdx.x;
  if (i < n8) {
    v4f x0 = *(const v4f*)(src + (size_t)i * 8);
    v4f x1 = *(const v4f*)(src + (size_t)i * 8 + 4);
    v8h y;
#pragma unroll
    for (int t = 0; t < 4; ++t) { y[t] = (_Float16)x0[t]; y[4 + t] = (_Float16)x1[t]; }
    *(v8h*)(dst + (size_t)i * 8) = y;
  }
}

// ---------------- NT GEMM: C[m,n] = (sum_k X[m,k]*W16[n,k] + bias[n]) * outscale ----
// One wave computes a 32x64 tile (2 A operands, 8 accumulators); all 8 waves
// of a block share the same 32-row A panel, which is staged into LDS once per
// k-step with async global->LDS DMA (ASYNCcnt), then read back as ds_load.
// MODE 0: store f16 head-split            dst[((b*H+h)*S + s)*DK + dk]
// MODE 1: store f16 head-split transposed dst[((b*H+h)*DK + dk)*S + s]  (for V)
// MODE 2: store f32 plain                 dst[m*D + n]                  (output)
template <typename TX, int MODE>
__global__ void __launch_bounds__(256)
gemm_nt(const TX* __restrict__ X, const _Float16* __restrict__ W,
        const float* __restrict__ bias, float outscale, void* __restrict__ out) {
  __shared__ __align__(16) TX atile[32 * 32];   // 4 KB (f32) or 2 KB (f16)

  const int tid  = threadIdx.x;
  const int lane = tid & 31;
  const int wave = blockIdx.x * 8 + (tid >> 5);
  const int NT = DD / 64;               // 16 n-tiles of width 64
  const int mb = (wave / NT) * 32;      // uniform across the block
  const int nb = (wave % NT) * 64;

  // cooperative async-copy addressing: 32 rows x 32 cols tile, 256 threads
  const int arow = tid >> 3;            // 0..31
  const int ach  = tid & 7;             // 8 chunks per row

  v8f c[2][4] = {};
  for (int k = 0; k < DD; k += 32) {
    __syncthreads();                    // previous iteration done reading atile
    if constexpr (sizeof(TX) == 4) {    // f32: 16 B per thread (b128)
      const void* g = (const char*)(X + (size_t)(mb + arow) * DD + k) + ach * 16;
      unsigned l = (unsigned)(uintptr_t)((char*)atile + arow * 32 * 4 + ach * 16);
      async_copy_b128(l, g);
    } else {                            // f16: 8 B per thread (b64)
      const void* g = (const char*)(X + (size_t)(mb + arow) * DD + k) + ach * 8;
      unsigned l = (unsigned)(uintptr_t)((char*)atile + arow * 32 * 2 + ach * 8);
      async_copy_b64(l, g);
    }
    wait_async0();                      // my lanes' DMA complete
    __syncthreads();                    // whole tile visible to all waves

    v16h a0 = load_a((const TX*)atile, 32, 0,  0, lane);
    v16h a1 = load_a((const TX*)atile, 32, 16, 0, lane);
#pragma unroll
    for (int j = 0; j < 4; ++j) {
      v16h bj = load_b(W, DD, k, nb + j * 16, lane);
      c[0][j] = wmma16(a0, bj, c[0][j]);
      c[1][j] = wmma16(a1, bj, c[1][j]);
    }
  }

  const int hf = lane >> 4;
  const int cl = lane & 15;
#pragma unroll
  for (int i = 0; i < 2; ++i) {
#pragma unroll
    for (int j = 0; j < 4; ++j) {
#pragma unroll
      for (int r = 0; r < 8; ++r) {
        int m = mb + i * 16 + r + 8 * hf;
        int n = nb + j * 16 + cl;
        float v = (c[i][j][r] + bias[n]) * outscale;
        if (MODE == 0) {
          int b_ = m / SS, s = m % SS, h = n / DKK, dk = n % DKK;
          ((_Float16*)out)[(((size_t)(b_ * HH + h) * SS) + s) * DKK + dk] = (_Float16)v;
        } else if (MODE == 1) {
          int b_ = m / SS, s = m % SS, h = n / DKK, dk = n % DKK;
          ((_Float16*)out)[(((size_t)(b_ * HH + h) * DKK) + dk) * SS + s] = (_Float16)v;
        } else {
          ((float*)out)[(size_t)m * DD + n] = v;
        }
      }
    }
  }
}

// ---------------- Flash attention (causal), one wave per 16-row tile ----------------
// Qh (pre-scaled by 1/sqrt(DK)), Kh: [B,H,S,DK] f16 ; Vt: [B,H,DK,S] f16 ;
// ctx out: [B*S, D] f16
__global__ void __launch_bounds__(256)
flash_attn(const _Float16* __restrict__ Qh, const _Float16* __restrict__ Kh,
           const _Float16* __restrict__ Vt, _Float16* __restrict__ ctx) {
  __shared__ __align__(16) _Float16 plds[8 * 16 * 32];   // 8 waves x (16x32) P tile

  const int lane  = threadIdx.x & 31;
  const int wslot = threadIdx.x >> 5;
  const int wave  = blockIdx.x * 8 + wslot;
  const int MT = SS / 16;                 // 128 m-tiles per (b,h)
  const int mt = wave & (MT - 1);
  const int h  = (wave >> 7) & (HH - 1);
  const int b  = wave >> 11;

  const _Float16* Qb = Qh + (size_t)(b * HH + h) * SS * DKK;
  const _Float16* Kb = Kh + (size_t)(b * HH + h) * SS * DKK;
  const _Float16* Vb = Vt + (size_t)(b * HH + h) * DKK * SS;
  _Float16* myp = plds + wslot * (16 * 32);

  const int mb = mt * 16;
  const int hf = lane >> 4;
  const int cl = lane & 15;

  // Q tile 16x64 -> two A operands (K = d)
  v16h aq0 = load_a(Qb, DKK, mb, 0,  lane);
  v16h aq1 = load_a(Qb, DKK, mb, 32, lane);

  v8f acc0 = {}, acc1 = {}, acc2 = {}, acc3 = {};
  float mi[8], li[8];
#pragma unroll
  for (int r = 0; r < 8; ++r) { mi[r] = -3.0e38f; li[r] = 0.0f; }

  for (int tb = 0; tb < mb + 16; tb += 32) {
    // ---- scores: S = Q K^T  (two 16x16 tiles over 32 t-columns) ----
    v8f s0 = {}, s1 = {};
    {
      v16h bk;
      bk = load_b(Kb, DKK, 0,  tb,      lane); s0 = wmma16(aq0, bk, s0);
      bk = load_b(Kb, DKK, 32, tb,      lane); s0 = wmma16(aq1, bk, s0);
      bk = load_b(Kb, DKK, 0,  tb + 16, lane); s1 = wmma16(aq0, bk, s1);
      bk = load_b(Kb, DKK, 32, tb + 16, lane); s1 = wmma16(aq1, bk, s1);
    }

    // ---- causal mask (-1e9 like the reference; scale pre-folded into Q) ----
    float p0[8], p1[8], mx[8];
    const int t0 = tb + cl, t1 = tb + 16 + cl;
#pragma unroll
    for (int r = 0; r < 8; ++r) {
      int m = mb + r + 8 * hf;
      float v0 = (t0 <= m) ? s0[r] : -1.0e9f;
      float v1 = (t1 <= m) ? s1[r] : -1.0e9f;
      p0[r] = v0; p1[r] = v1;
      mx[r] = fmaxf(v0, v1);
    }
    // row max across the 16 lanes of each half-wave
#pragma unroll
    for (int off = 1; off < 16; off <<= 1) {
#pragma unroll
      for (int r = 0; r < 8; ++r)
        mx[r] = fmaxf(mx[r], __shfl_xor(mx[r], off, 32));
    }

    // ---- online softmax update ----
    float sc[8], rs[8];
#pragma unroll
    for (int r = 0; r < 8; ++r) {
      float nm = fmaxf(mi[r], mx[r]);
      sc[r] = __expf(mi[r] - nm);
      mi[r] = nm;
      p0[r] = __expf(p0[r] - nm);
      p1[r] = __expf(p1[r] - nm);
      rs[r] = p0[r] + p1[r];
    }
#pragma unroll
    for (int off = 1; off < 16; off <<= 1) {
#pragma unroll
      for (int r = 0; r < 8; ++r)
        rs[r] += __shfl_xor(rs[r], off, 32);
    }
#pragma unroll
    for (int r = 0; r < 8; ++r) {
      li[r] = li[r] * sc[r] + rs[r];
      acc0[r] *= sc[r]; acc1[r] *= sc[r]; acc2[r] *= sc[r]; acc3[r] *= sc[r];
    }

    // ---- re-layout P (C-format -> A-format) through LDS ----
#pragma unroll
    for (int r = 0; r < 8; ++r) {
      int row = r + 8 * hf;
      myp[row * 32 + cl]      = (_Float16)p0[r];
      myp[row * 32 + 16 + cl] = (_Float16)p1[r];
    }
    v16h ap = load_a((const _Float16*)myp, 32, 0, 0, lane);   // same-wave LDS RAW: in-order DS pipe

    // ---- ctx += P @ V  (V transposed: B(k=t, n=d) = Vt[d, t], t contiguous) ----
    {
      v16h bv;
      bv = load_b(Vb, SS, tb, 0,  lane); acc0 = wmma16(ap, bv, acc0);
      bv = load_b(Vb, SS, tb, 16, lane); acc1 = wmma16(ap, bv, acc1);
      bv = load_b(Vb, SS, tb, 32, lane); acc2 = wmma16(ap, bv, acc2);
      bv = load_b(Vb, SS, tb, 48, lane); acc3 = wmma16(ap, bv, acc3);
    }
  }

  // ---- normalize and store ctx tile (merge heads: [B*S, D]) ----
#pragma unroll
  for (int r = 0; r < 8; ++r) {
    float inv = 1.0f / li[r];
    int s = mb + r + 8 * hf;
    size_t base = ((size_t)b * SS + s) * DD + h * DKK;
    ctx[base + 0  + cl] = (_Float16)(acc0[r] * inv);
    ctx[base + 16 + cl] = (_Float16)(acc1[r] * inv);
    ctx[base + 32 + cl] = (_Float16)(acc2[r] * inv);
    ctx[base + 48 + cl] = (_Float16)(acc3[r] * inv);
  }
}

// ---------------- host-side launch ----------------
extern "C" void kernel_launch(void* const* d_in, const int* in_sizes, int n_in,
                              void* d_out, int out_size, void* d_ws, size_t ws_size,
                              hipStream_t stream) {
  const float* q  = (const float*)d_in[0];
  const float* k  = (const float*)d_in[1];
  const float* v  = (const float*)d_in[2];
  // d_in[3] = mask (int, causal) — known statically, not needed on device
  const float* Wq = (const float*)d_in[4];
  const float* bq = (const float*)d_in[5];
  const float* Wk = (const float*)d_in[6];
  const float* bk = (const float*)d_in[7];
  const float* Wv = (const float*)d_in[8];
  const float* bv = (const float*)d_in[9];
  const float* Wo = (const float*)d_in[10];
  const float* bo = (const float*)d_in[11];
  float* out = (float*)d_out;

  // workspace: Qh | Kh | Vt | ctx (each 8Mi f16 = 16 MB) then 4x f16 weights (2 MB each)
  const size_t HEAD_ELEMS = (size_t)BB * HH * SS * DKK;   // 8388608
  const size_t W_ELEMS    = (size_t)DD * DD;              // 1048576
  _Float16* Qh   = (_Float16*)d_ws;
  _Float16* Kh   = Qh  + HEAD_ELEMS;
  _Float16* Vt   = Kh  + HEAD_ELEMS;
  _Float16* ctx  = Vt  + HEAD_ELEMS;
  _Float16* Wq16 = ctx + HEAD_ELEMS;
  _Float16* Wk16 = Wq16 + W_ELEMS;
  _Float16* Wv16 = Wk16 + W_ELEMS;
  _Float16* Wo16 = Wv16 + W_ELEMS;

  // ---- prepass: convert the 4 weight matrices to f16 (once, ~8 MB total) ----
  {
    int n8 = (int)(W_ELEMS / 8);                 // 131072 threads
    dim3 g((n8 + 255) / 256), bl(256);
    cvt_f32_f16<<<g, bl, 0, stream>>>(Wq, Wq16, n8);
    cvt_f32_f16<<<g, bl, 0, stream>>>(Wk, Wk16, n8);
    cvt_f32_f16<<<g, bl, 0, stream>>>(Wv, Wv16, n8);
    cvt_f32_f16<<<g, bl, 0, stream>>>(Wo, Wo16, n8);
  }

  // (M/32)*(D/64) = 256*16 = 4096 waves -> 512 blocks of 8 waves
  dim3 ggemm(512), block(256);
  const float qscale = 0.125f;   // 1/sqrt(DK) folded into Q projection

  gemm_nt<float, 0><<<ggemm, block, 0, stream>>>(q, Wq16, bq, qscale, Qh);
  gemm_nt<float, 0><<<ggemm, block, 0, stream>>>(k, Wk16, bk, 1.0f,  Kh);
  gemm_nt<float, 1><<<ggemm, block, 0, stream>>>(v, Wv16, bv, 1.0f,  Vt);

  // B*H*(S/16) = 8192 waves -> 1024 blocks
  dim3 gattn(1024);
  flash_attn<<<gattn, block, 0, stream>>>(Qh, Kh, Vt, ctx);

  gemm_nt<_Float16, 2><<<ggemm, block, 0, stream>>>(ctx, Wo16, bo, 1.0f, (void*)out);
}
